// FraudDetectionGNN_55422257988059
// MI455X (gfx1250) — matrix operand order repaired and checked by reading, not compiled, not run on hardware
//
#include <hip/hip_runtime.h>
#include <math.h>

// ---------------------------------------------------------------- constants
#define NN   50000          // nodes  (50000 = 3125*16, tile-exact)
#define NE   1200000        // edges  (1200000 = 75000*16)
#define EAUG (NE + NN)      // edges + self loops
#define FIN  128
#define FE   16
#define CHX  64             // per-head channels
#define H1   4              // heads layer 1
#define SLOPE 0.2f

typedef _Float16 v16h __attribute__((ext_vector_type(16)));
typedef float    v8f  __attribute__((ext_vector_type(8)));
typedef int      vec4i __attribute__((vector_size(16)));   // matches builtin param type

#if __has_builtin(__builtin_amdgcn_global_load_async_to_lds_b128) && \
    __has_builtin(__builtin_amdgcn_s_wait_asynccnt)
#define HAVE_ASYNC_LDS 1
typedef __attribute__((address_space(1))) vec4i g_vec4i;   // global
typedef __attribute__((address_space(3))) vec4i l_vec4i;   // LDS
#else
#define HAVE_ASYNC_LDS 0
#endif

// float atomic max via ordered-int mapping (works for mixed signs)
__device__ __forceinline__ void atomicMaxF(float* a, float v) {
  if (v >= 0.0f) atomicMax((int*)a, __float_as_int(v));
  else           atomicMin((unsigned int*)a, __float_as_uint(v));
}

// convert 8 consecutive f32 -> halves [off..off+7] of a v16h (2x b128 loads)
__device__ __forceinline__ void cvt8(v16h& a, int off, const float* p) {
  float4 lo = *(const float4*)p;
  float4 hi = *(const float4*)(p + 4);
  a[off + 0] = (_Float16)lo.x; a[off + 1] = (_Float16)lo.y;
  a[off + 2] = (_Float16)lo.z; a[off + 3] = (_Float16)lo.w;
  a[off + 4] = (_Float16)hi.x; a[off + 5] = (_Float16)hi.y;
  a[off + 6] = (_Float16)hi.z; a[off + 7] = (_Float16)hi.w;
}

__device__ __forceinline__ void cvt16(v16h& b, const float* p) {
  cvt8(b, 0, p);
  cvt8(b, 8, p + 8);
}

// ---------------------------------------------------------------- utility
__global__ void k_fill(float* p, float v, long n) {
  long i = (long)blockIdx.x * blockDim.x + threadIdx.x;
  if (i < n) p[i] = v;
}

// ---------------------------------------------------------------- self loops
__global__ void k_loop_accum(const int* __restrict__ dst,
                             const float* __restrict__ eattr,
                             float* __restrict__ asum, float* __restrict__ cnt) {
  int e = blockIdx.x * blockDim.x + threadIdx.x;
  if (e >= NE) return;
  int d = dst[e];
  atomicAdd(&cnt[d], 1.0f);
  const float* a = eattr + (size_t)e * FE;
  #pragma unroll
  for (int f = 0; f < FE; ++f) atomicAdd(&asum[(size_t)d * FE + f], a[f]);
}

__global__ void k_loop_final(float* __restrict__ asum, const float* __restrict__ cnt) {
  long i = (long)blockIdx.x * blockDim.x + threadIdx.x;
  if (i >= (long)NN * FE) return;
  int n = (int)(i / FE);
  asum[i] = asum[i] / fmaxf(cnt[n], 1.0f);
}

// ---------------------------------------------------------------- fold We @ attE
// Ae[f,h] = sum_c We[f, h*CHX + c] * attE[h, c]   (tiny, one block)
__global__ void k_fold_ae(const float* __restrict__ We, const float* __restrict__ attE,
                          float* __restrict__ Ae, int heads) {
  int f = threadIdx.x, h = threadIdx.y;
  if (f >= FE || h >= heads) return;
  float s = 0.f;
  for (int c = 0; c < CHX; ++c) s += We[(size_t)f * heads * CHX + h * CHX + c] * attE[h * CHX + c];
  Ae[f * heads + h] = s;
}

// ---------------------------------------------------------------- WMMA GEMM
// C[M,N] = A[M,K] * B[K,N]; f32 in/out, f16 multiply, f32 accumulate.
// Requires M % 16 == 0, N % 16 == 0, K % 32 == 0 (all true here).
// One wave computes one 16x16 tile of C; K fully unrolled at compile time.
template <int K>
__global__ __launch_bounds__(128)
void k_wmma_gemm(const float* __restrict__ A, const float* __restrict__ B,
                 float* __restrict__ C, int M, int N) {
  int wave = blockIdx.x * (blockDim.x >> 5) + (threadIdx.x >> 5);
  int lane = threadIdx.x & 31;
  int ntiles = N >> 4;
  int mt = wave / ntiles;
  int nt = wave - mt * ntiles;
  if (mt * 16 >= M) return;                       // wave-uniform exit

  int arow  = mt * 16 + (lane & 15);
  int kbase = (lane >> 4) * 8;                    // ISA A-layout: lanes 16-31 -> K+8
  const float* Arow  = A + (size_t)arow * K;
  const float* Bbase = B + (size_t)nt * 16;

  v8f acc = {};
  #pragma unroll
  for (int k0 = 0; k0 < K; k0 += 32) {
    v16h a;
    cvt8(a, 0, Arow + k0 + kbase);
    cvt8(a, 8, Arow + k0 + kbase + 16);
    v16h b;
    #pragma unroll
    for (int s = 0; s < 16; ++s)                  // B col = lane%16, K in table order
      b[s] = (_Float16)Bbase[(size_t)(k0 + kbase + (s < 8 ? s : s + 8)) * N + (lane & 15)];
    acc = __builtin_amdgcn_wmma_f32_16x16x32_f16(false, a, false, b,
                                                 (short)0, acc, false, false);
  }
  // D layout: c[i] -> row mt*16 + i + 8*(lane/16), col nt*16 + lane%16
  int mlo = mt * 16 + 8 * (lane >> 4);
  int col = nt * 16 + (lane & 15);
  #pragma unroll
  for (int i = 0; i < 8; ++i) C[(size_t)(mlo + i) * N + col] = acc[i];
}

// ---------------------------------------------------------------- node attention terms
__global__ void k_node_att(const float* __restrict__ hpre,
                           const float* __restrict__ attS, const float* __restrict__ attD,
                           float* __restrict__ aS, float* __restrict__ aD, int heads) {
  long i = (long)blockIdx.x * blockDim.x + threadIdx.x;
  if (i >= (long)NN * heads) return;
  int n = (int)(i / heads), h = (int)(i % heads);
  const float* hp = hpre + (size_t)n * heads * CHX + h * CHX;
  float s = 0.f, d = 0.f;
  for (int c = 0; c < CHX; ++c) { s += hp[c] * attS[h * CHX + c]; d += hp[c] * attD[h * CHX + c]; }
  aS[i] = s; aD[i] = d;
}

// ---------------------------------------------------------------- attention passes
__global__ void k_logits_max(const int* __restrict__ src, const int* __restrict__ dst,
                             const float* __restrict__ eattr, const float* __restrict__ lattr,
                             const float* __restrict__ Ae,
                             const float* __restrict__ aS, const float* __restrict__ aD,
                             float* __restrict__ logit, float* __restrict__ m, int heads) {
  int e = blockIdx.x * blockDim.x + threadIdx.x;
  if (e >= EAUG) return;
  int s, d; const float* at;
  if (e < NE) { s = src[e]; d = dst[e]; at = eattr + (size_t)e * FE; }
  else        { s = d = e - NE;         at = lattr + (size_t)(e - NE) * FE; }
  for (int h = 0; h < heads; ++h) {
    float ae = 0.f;
    #pragma unroll
    for (int f = 0; f < FE; ++f) ae += at[f] * Ae[f * heads + h];
    float l = aS[(size_t)s * heads + h] + aD[(size_t)d * heads + h] + ae;
    l = (l > 0.f) ? l : SLOPE * l;
    logit[(size_t)e * heads + h] = l;
    atomicMaxF(&m[(size_t)d * heads + h], l);
  }
}

__global__ void k_expsum(const int* __restrict__ dst, const float* __restrict__ logit,
                         const float* __restrict__ m, float* __restrict__ ssum, int heads) {
  int e = blockIdx.x * blockDim.x + threadIdx.x;
  if (e >= EAUG) return;
  int d = (e < NE) ? dst[e] : (e - NE);
  for (int h = 0; h < heads; ++h) {
    float ex = __expf(logit[(size_t)e * heads + h] - m[(size_t)d * heads + h]);
    atomicAdd(&ssum[(size_t)d * heads + h], ex);
  }
}

__global__ void k_scatter(const int* __restrict__ src, const int* __restrict__ dst,
                          const float* __restrict__ logit, const float* __restrict__ m,
                          const float* __restrict__ ssum, const float* __restrict__ hpre,
                          float* __restrict__ out, int heads) {
  long idx = (long)blockIdx.x * blockDim.x + threadIdx.x;
  if (idx >= (long)EAUG * heads) return;
  int e = (int)(idx / heads), h = (int)(idx % heads);
  int s, d;
  if (e < NE) { s = src[e]; d = dst[e]; } else s = d = e - NE;
  float w = __expf(logit[idx] - m[(size_t)d * heads + h]) / ssum[(size_t)d * heads + h];
  const float* hp = hpre + (size_t)s * heads * CHX + h * CHX;
  float*       op = out  + (size_t)d * heads * CHX + h * CHX;
  #pragma unroll 8
  for (int c = 0; c < CHX; ++c) atomicAdd(&op[c], w * hp[c]);
}

__global__ void k_bias_relu(float* __restrict__ x, const float* __restrict__ b, long rows, int F) {
  long i = (long)blockIdx.x * blockDim.x + threadIdx.x;
  if (i >= rows * F) return;
  float v = x[i] + b[i % F];
  x[i] = v > 0.f ? v : 0.f;
}

// ---------------------------------------------------------------- edge classifier
// z = relu([h2[s] | h2[d] | eattr] @ Wc1 + bc1); out = sigmoid(z @ Wc2 + bc2)
// K = 144 padded to 160; WMMA f16; one wave -> 16 edges; 8 waves / block.
// Wc1 is staged once per block into LDS (zero-padded to 160 rows) via
// GLOBAL_LOAD_ASYNC_TO_LDS_B128 (+ s_wait_asynccnt) when available; the
// WMMA loop is branch-free and reads B as single f32 LDS loads.
__global__ __launch_bounds__(256)
void k_edge_mlp(const int* __restrict__ src, const int* __restrict__ dst,
                const float* __restrict__ eattr, const float* __restrict__ h2,
                const float* __restrict__ Wc1, const float* __restrict__ bc1,
                const float* __restrict__ Wc2, const float* __restrict__ bc2,
                float* __restrict__ out) {
  __shared__ float    wlds[160 * 64];     // 40 KB  (B panel, f32)
  __shared__ _Float16 zbuf[8][16][64];    // 16 KB  (z tiles, f16)
  int tid  = threadIdx.x;
  int w    = tid >> 5;
  int lane = tid & 31;
  int e0   = (blockIdx.x * 8 + w) * 16;

  // ---- stage Wc1 (144x64 f32) -> LDS, pad rows 144..159 with zeros ----
#if HAVE_ASYNC_LDS
  for (int i = tid; i < (144 * 64 / 4); i += 256) {
    int off = i * 4;
    __builtin_amdgcn_global_load_async_to_lds_b128(
        (g_vec4i*)(Wc1 + off), (l_vec4i*)&wlds[off], 0, 0);
  }
  for (int i = tid; i < (16 * 64 / 4); i += 256) {
    *(float4*)&wlds[144 * 64 + i * 4] = make_float4(0.f, 0.f, 0.f, 0.f);
  }
  __builtin_amdgcn_s_wait_asynccnt(0);
  __syncthreads();
#else
  for (int i = tid; i < (160 * 64 / 4); i += 256) {
    int off = i * 4;
    float4 v = (off < 144 * 64) ? *(const float4*)(Wc1 + off)
                                : make_float4(0.f, 0.f, 0.f, 0.f);
    *(float4*)&wlds[off] = v;
  }
  __syncthreads();
#endif

  // ---- gather per-edge feature rows as A fragments (static segment map) ----
  int er = e0 + (lane & 15);
  int ee = er < NE ? er : NE - 1;
  int s = src[ee], d = dst[ee];
  int kbase = (lane >> 4) * 8;

  const float* hs  = h2 + (size_t)s * CHX + kbase;
  const float* hd  = h2 + (size_t)d * CHX + kbase;
  const float* eap = eattr + (size_t)ee * FE + kbase;

  v16h afr[5];
  cvt8(afr[0], 0, hs);       cvt8(afr[0], 8, hs + 16);
  cvt8(afr[1], 0, hs + 32);  cvt8(afr[1], 8, hs + 48);
  cvt8(afr[2], 0, hd);       cvt8(afr[2], 8, hd + 16);
  cvt8(afr[3], 0, hd + 32);  cvt8(afr[3], 8, hd + 48);
  cvt8(afr[4], 0, eap);
  #pragma unroll
  for (int e = 8; e < 16; ++e) afr[4][e] = (_Float16)0.f;

  // ---- WMMA: 4 N-tiles x 5 K-steps, B fragments from LDS (branch-free) ----
  v8f acc[4] = {};
  #pragma unroll
  for (int nt = 0; nt < 4; ++nt) {
    #pragma unroll
    for (int ks = 0; ks < 5; ++ks) {
      int kk = ks * 32 + kbase;              // B col = lane%16, K in table order
      v16h b;
      #pragma unroll
      for (int s = 0; s < 16; ++s)
        b[s] = (_Float16)wlds[(kk + (s < 8 ? s : s + 8)) * 64 + nt * 16 + (lane & 15)];
      acc[nt] = __builtin_amdgcn_wmma_f32_16x16x32_f16(false, afr[ks], false, b,
                                                       (short)0, acc[nt], false, false);
    }
  }

  // ---- z tile -> LDS (f16) with bias + relu ----
  int rbase = 8 * (lane >> 4);
  int cl    = lane & 15;
  #pragma unroll
  for (int nt = 0; nt < 4; ++nt) {
    float bb = bc1[nt * 16 + cl];
    #pragma unroll
    for (int i = 0; i < 8; ++i) {
      float v = acc[nt][i] + bb;
      zbuf[w][rbase + i][nt * 16 + cl] = (_Float16)(v > 0.f ? v : 0.f);
    }
  }
  __syncthreads();

  // ---- out = sigmoid(z . Wc2 + bc2) ----
  if (lane < 16) {
    int erow = e0 + lane;
    float sdot = bc2[0];
    #pragma unroll 8
    for (int n = 0; n < 64; ++n) sdot += (float)zbuf[w][lane][n] * Wc2[n];
    if (erow < NE) out[erow] = 1.0f / (1.0f + __expf(-sdot));
  }
}

// ---------------------------------------------------------------- launch
extern "C" void kernel_launch(void* const* d_in, const int* in_sizes, int n_in,
                              void* d_out, int out_size, void* d_ws, size_t ws_size,
                              hipStream_t stream) {
  const float* x     = (const float*)d_in[0];
  const int*   ei    = (const int*)  d_in[1];
  const float* eattr = (const float*)d_in[2];
  const float* W1    = (const float*)d_in[3];
  const float* attS1 = (const float*)d_in[4];
  const float* attD1 = (const float*)d_in[5];
  const float* We1   = (const float*)d_in[6];
  const float* attE1 = (const float*)d_in[7];
  const float* b1    = (const float*)d_in[8];
  const float* W2    = (const float*)d_in[9];
  const float* attS2 = (const float*)d_in[10];
  const float* attD2 = (const float*)d_in[11];
  const float* We2   = (const float*)d_in[12];
  const float* attE2 = (const float*)d_in[13];
  const float* b2    = (const float*)d_in[14];
  const float* Wc1   = (const float*)d_in[15];
  const float* bc1   = (const float*)d_in[16];
  const float* Wc2   = (const float*)d_in[17];
  const float* bc2   = (const float*)d_in[18];
  float* outp = (float*)d_out;
  const int* src = ei;
  const int* dst = ei + NE;

  // workspace carve-up (all chunk sizes are multiples of 4 floats -> 16B aligned)
  float* p = (float*)d_ws;
  float* cnt   = p; p += NN;
  float* lattr = p; p += (size_t)NN * FE;       // self-loop attrs (mean)
  float* Ae1   = p; p += FE * H1;
  float* Ae2   = p; p += FE;
  float* h1pre = p; p += (size_t)NN * H1 * CHX; // N x 256
  float* aS1   = p; p += (size_t)NN * H1;
  float* aD1   = p; p += (size_t)NN * H1;
  float* lg1   = p; p += (size_t)EAUG * H1;
  float* m1    = p; p += (size_t)NN * H1;
  float* s1    = p; p += (size_t)NN * H1;
  float* out1  = p; p += (size_t)NN * H1 * CHX; // N x 256
  float* h2pre = p; p += (size_t)NN * CHX;      // N x 64
  float* aS2   = p; p += NN;
  float* aD2   = p; p += NN;
  float* lg2   = p; p += (size_t)EAUG;
  float* m2    = p; p += NN;
  float* s2    = p; p += NN;
  float* out2  = p; p += (size_t)NN * CHX;      // N x 64

  auto blocks = [](long n, int b) { return (unsigned)((n + b - 1) / b); };
  const float NINF = -__builtin_huge_valf();

  // self loops: mean of incoming edge attrs
  k_fill<<<blocks(NN, 256), 256, 0, stream>>>(cnt, 0.f, NN);
  k_fill<<<blocks((long)NN * FE, 256), 256, 0, stream>>>(lattr, 0.f, (long)NN * FE);
  k_loop_accum<<<blocks(NE, 256), 256, 0, stream>>>(dst, eattr, lattr, cnt);
  k_loop_final<<<blocks((long)NN * FE, 256), 256, 0, stream>>>(lattr, cnt);

  // fold edge-attention weights
  k_fold_ae<<<1, dim3(16, H1), 0, stream>>>(We1, attE1, Ae1, H1);
  k_fold_ae<<<1, dim3(16, 1), 0, stream>>>(We2, attE2, Ae2, 1);

  // ---- layer 1 ----
  {
    long waves = (long)(NN / 16) * (H1 * CHX / 16);
    k_wmma_gemm<FIN><<<blocks(waves, 4), 128, 0, stream>>>(x, W1, h1pre, NN, H1 * CHX);
  }
  k_node_att<<<blocks((long)NN * H1, 256), 256, 0, stream>>>(h1pre, attS1, attD1, aS1, aD1, H1);
  k_fill<<<blocks((long)NN * H1, 256), 256, 0, stream>>>(m1, NINF, (long)NN * H1);
  k_fill<<<blocks((long)NN * H1, 256), 256, 0, stream>>>(s1, 0.f, (long)NN * H1);
  k_fill<<<blocks((long)NN * H1 * CHX, 256), 256, 0, stream>>>(out1, 0.f, (long)NN * H1 * CHX);
  k_logits_max<<<blocks(EAUG, 256), 256, 0, stream>>>(src, dst, eattr, lattr, Ae1, aS1, aD1, lg1, m1, H1);
  k_expsum<<<blocks(EAUG, 256), 256, 0, stream>>>(dst, lg1, m1, s1, H1);
  k_scatter<<<blocks((long)EAUG * H1, 256), 256, 0, stream>>>(src, dst, lg1, m1, s1, h1pre, out1, H1);
  k_bias_relu<<<blocks((long)NN * H1 * CHX, 256), 256, 0, stream>>>(out1, b1, NN, H1 * CHX);

  // ---- layer 2 ----
  {
    long waves = (long)(NN / 16) * (CHX / 16);
    k_wmma_gemm<H1 * CHX><<<blocks(waves, 4), 128, 0, stream>>>(out1, W2, h2pre, NN, CHX);
  }
  k_node_att<<<blocks(NN, 256), 256, 0, stream>>>(h2pre, attS2, attD2, aS2, aD2, 1);
  k_fill<<<blocks(NN, 256), 256, 0, stream>>>(m2, NINF, NN);
  k_fill<<<blocks(NN, 256), 256, 0, stream>>>(s2, 0.f, NN);
  k_fill<<<blocks((long)NN * CHX, 256), 256, 0, stream>>>(out2, 0.f, (long)NN * CHX);
  k_logits_max<<<blocks(EAUG, 256), 256, 0, stream>>>(src, dst, eattr, lattr, Ae2, aS2, aD2, lg2, m2, 1);
  k_expsum<<<blocks(EAUG, 256), 256, 0, stream>>>(dst, lg2, m2, s2, 1);
  k_scatter<<<blocks(EAUG, 256), 256, 0, stream>>>(src, dst, lg2, m2, s2, h2pre, out2, 1);
  k_bias_relu<<<blocks((long)NN * CHX, 256), 256, 0, stream>>>(out2, b2, NN, CHX);

  // ---- edge classifier (WMMA, async-LDS-staged weights, fused sigmoid) ----
  k_edge_mlp<<<blocks(NE, 128), 256, 0, stream>>>(src, dst, eattr, out2, Wc1, bc1, Wc2, bc2, outp);

  (void)in_sizes; (void)n_in; (void)out_size; (void)ws_size;
}
